// FlashLinearAttention_8126078124549
// MI455X (gfx1250) — matrix operand order repaired and checked by reading, not compiled, not run on hardware
//
#include <hip/hip_runtime.h>

#define BATCH 4
#define EDIM 1024
#define SDIM 128
#define EHDIM 2048
#define CHUNKW 64
#define NCH 32
#define LP 2048
#define ROWS (BATCH * LP)

typedef __attribute__((ext_vector_type(16))) __bf16 bf16x16;
typedef __attribute__((ext_vector_type(8)))  float  f32x8;
typedef unsigned int u32x4 __attribute__((ext_vector_type(4)));
typedef unsigned int uint32x4_t __attribute__((ext_vector_type(4)));
typedef int int32x4_t __attribute__((ext_vector_type(4)));
typedef int int32x8_t __attribute__((ext_vector_type(8)));

struct FragAB { union { bf16x16 v; u32x4 q[2]; unsigned short h[16]; }; };

__device__ __forceinline__ unsigned short f32_to_bf16_bits(float f) {
  union { float f; unsigned u; } c; c.f = f;
  unsigned r = c.u + 0x7FFFu + ((c.u >> 16) & 1u);
  return (unsigned short)(r >> 16);
}
__device__ __forceinline__ float bf16_bits_to_f32(unsigned short hb) {
  union { unsigned u; float f; } c; c.u = ((unsigned)hb) << 16;
  return c.f;
}
__device__ __forceinline__ float fast_silu(float x) {
  // x * rcp(1+e^-x): v_exp + v_rcp instead of the IEEE div sequence
  return x * __builtin_amdgcn_rcpf(1.0f + __expf(-x));
}

__device__ __forceinline__ f32x8 wmma_bf16(bf16x16 a, bf16x16 b, f32x8 c) {
  // D = A(16x32) * B(32x16) + C, f32 accumulate
  return __builtin_amdgcn_wmma_f32_16x16x32_bf16(false, a, false, b, (short)0, c, false, false);
}

// A fragment: 16x32 bf16 tile at (row0,k0) from row-major [*, ld] bf16 matrix.
// Contiguous K order (same as load_B_frag): lane L -> M=L%16; K strip [16*(L/16), +16).
__device__ __forceinline__ bf16x16 load_A_frag(const unsigned short* X, int ld, int row0, int k0, int lane) {
  int m  = row0 + (lane & 15);
  int kb = (lane >> 4) << 4;
  const unsigned short* p = X + (size_t)m * ld + k0;
  FragAB f;
  f.q[0] = *(const u32x4*)(p + kb);
  f.q[1] = *(const u32x4*)(p + kb + 8);
  return f.v;
}

// B fragment: B[k][n] = W[n0+n][k0+k] from row-major [N, ld] (K-contiguous per column).
// Contiguous K order (same as load_A_frag): lane L -> N=L%16; K strip [16*(L/16), +16).
__device__ __forceinline__ bf16x16 load_B_frag(const unsigned short* W, int ld, int n0, int k0, int lane) {
  int n  = n0 + (lane & 15);
  int kb = (lane >> 4) << 4;
  const unsigned short* p = W + (size_t)n * ld + k0 + kb;
  FragAB f;
  f.q[0] = *(const u32x4*)(p);
  f.q[1] = *(const u32x4*)(p + 8);
  return f.v;
}

// ---------------- TDM: 1-D contiguous global -> LDS copy (nbytes multiple of 8) ----------------
__device__ __forceinline__ void tdm_copy_1d_to_lds(const void* gsrc, void* lds_dst, unsigned nbytes) {
  unsigned long long ga = (unsigned long long)(size_t)gsrc;
  unsigned lds = (unsigned)(size_t)lds_dst;   // low 32 bits of generic LDS addr = LDS offset
  unsigned n8 = nbytes >> 3;                  // elements of 8 bytes (data_size code 3)
  uint32x4_t g0;
  g0[0] = 1u;                                          // count=1 user descriptor
  g0[1] = lds;                                         // lds_addr
  g0[2] = (unsigned)(ga & 0xFFFFFFFFu);                // global_addr[31:0]
  g0[3] = (unsigned)((ga >> 32) & 0x1FFFFFFu) | (2u << 30);  // global_addr[56:32] | type=2
  int32x8_t g1;
  g1[0] = (int)(3u << 16);                             // data_size=8B; no mask/iterate/pad
  g1[1] = (int)((n8 & 0xFFFFu) << 16);                 // tensor_dim0[15:0]
  g1[2] = (int)((n8 >> 16) | (1u << 16));              // tensor_dim0[31:16] | tensor_dim1=1
  g1[3] = (int)(n8 << 16);                             // tile_dim0 = n8
  g1[4] = 0;                                           // tile_dim1=0, tile_dim2=0
  g1[5] = (int)n8;                                     // tensor_dim0_stride[31:0]
  g1[6] = 0;
  g1[7] = 0;
  int32x4_t z4 = {0, 0, 0, 0};
#if defined(__clang_major__) && __clang_major__ >= 23
  int32x8_t z8 = {0, 0, 0, 0, 0, 0, 0, 0};
  __builtin_amdgcn_tensor_load_to_lds(g0, g1, z4, z4, z8, 0);
#else
  __builtin_amdgcn_tensor_load_to_lds(g0, g1, z4, z4, 0);
#endif
}

// ---------------- weight f32 -> bf16 ----------------
__global__ void cvt_bf16_kernel(const float* __restrict__ src, unsigned short* __restrict__ dst, int n) {
  int i = blockIdx.x * blockDim.x + threadIdx.x;
  if (i < n) dst[i] = f32_to_bf16_bits(src[i]);
}

// ---------------- LayerNorm: query[L,B,E] -> xn bf16 [b*LP+l, E] ----------------
__global__ void ln_kernel(const float* __restrict__ q, const float* __restrict__ g,
                          const float* __restrict__ be, unsigned short* __restrict__ xn) {
  int row = blockIdx.x;                 // b*LP + l
  int bb = row / LP, l = row % LP;
  const float* src = q + ((size_t)l * BATCH + bb) * EDIM;
  __shared__ float red0[8], red1[8];
  float s = 0.f, s2 = 0.f;
  for (int e = threadIdx.x; e < EDIM; e += blockDim.x) { float x = src[e]; s += x; s2 += x * x; }
  for (int o = 16; o > 0; o >>= 1) { s += __shfl_down(s, o); s2 += __shfl_down(s2, o); }
  int wave = threadIdx.x >> 5, lane = threadIdx.x & 31;
  if (lane == 0) { red0[wave] = s; red1[wave] = s2; }
  __syncthreads();
  if (threadIdx.x == 0) {
    float a = 0.f, c = 0.f;
    for (int i = 0; i < 8; ++i) { a += red0[i]; c += red1[i]; }
    red0[0] = a; red1[0] = c;
  }
  __syncthreads();
  float mean = red0[0] * (1.0f / EDIM);
  float var  = red1[0] * (1.0f / EDIM) - mean * mean;
  float inv  = rsqrtf(var + 1e-5f);
  unsigned short* dst = xn + (size_t)row * EDIM;
  for (int e = threadIdx.x; e < EDIM; e += blockDim.x)
    dst[e] = f32_to_bf16_bits((src[e] - mean) * inv * g[e] + be[e]);
}

// ---------------- GEMM1: silu(xn @ W^T + b); mode0 -> transposed bf16 [B][N][LP]; mode1 -> row-major f32 ----------------
__global__ void gemm1_kernel(const unsigned short* __restrict__ Xn,  // [ROWS][E] bf16
                             const unsigned short* __restrict__ Wt,  // [N][E] bf16
                             const float* __restrict__ bias, int Ntot, int mode,
                             unsigned short* __restrict__ dstT, float* __restrict__ dstRM) {
  int lane = threadIdx.x & 31, wave = threadIdx.x >> 5;
  int m0 = blockIdx.x * 64;
  int n0 = blockIdx.y * 128 + wave * 16;
  f32x8 acc[4] = {};
  for (int k = 0; k < EDIM; k += 32) {
    bf16x16 bfr = load_B_frag(Wt, EDIM, n0, k, lane);
    __builtin_prefetch(Wt + (size_t)(n0 + (lane & 15)) * EDIM + k + 128, 0, 0);
#pragma unroll
    for (int mt = 0; mt < 4; ++mt) {
      bf16x16 afr = load_A_frag(Xn, EDIM, m0 + mt * 16, k, lane);
      acc[mt] = wmma_bf16(afr, bfr, acc[mt]);
    }
  }
  int n = n0 + (lane & 15);
  float bs = bias[n];
#pragma unroll
  for (int mt = 0; mt < 4; ++mt) {
#pragma unroll
    for (int r = 0; r < 8; ++r) {
      int m = m0 + mt * 16 + ((lane >> 4) << 3) + r;
      float sv = fast_silu(acc[mt][r] + bs);
      if (mode == 0) {
        int bb = m / LP, l = m % LP;
        dstT[((size_t)bb * Ntot + n) * LP + l] = f32_to_bf16_bits(sv);
      } else {
        dstRM[(size_t)m * Ntot + n] = sv;
      }
    }
  }
}

// ---------------- RoPE + per-channel affine ----------------
__global__ void rope_kernel(const float* __restrict__ base,
                            const float* __restrict__ qqw, const float* __restrict__ qqb,
                            const float* __restrict__ qkw, const float* __restrict__ qkb,
                            const float* __restrict__ lqw, const float* __restrict__ lqb,
                            const float* __restrict__ lkw, const float* __restrict__ lkb,
                            unsigned short* __restrict__ quadq, unsigned short* __restrict__ quadk,
                            unsigned short* __restrict__ linq, unsigned short* __restrict__ linkT) {
  int id = blockIdx.x * blockDim.x + threadIdx.x;
  int row = id >> 6, j = id & 63;
  if (row >= ROWS) return;
  int bb = row / LP, l = row % LP;
  int g = l / CHUNKW, nn = l % CHUNKW;
  float ang = (float)l * __expf(-(float)j * 0.14391156947f); // ln(10000)/64
  float sn = __sinf(ang), cs = __cosf(ang);
  float b1 = base[(size_t)row * SDIM + j];
  float b2 = base[(size_t)row * SDIM + j + 64];
#define APPLY_ROPE(wv, bv, O1, O2)                                          \
  { float t1 = b1 * wv[j] + bv[j]; float t2 = b2 * wv[j + 64] + bv[j + 64]; \
    O1 = t1 * cs - t2 * sn; O2 = t2 * cs + t1 * sn; }
  float o1, o2;
  size_t rS = (size_t)row * SDIM;
  APPLY_ROPE(qqw, qqb, o1, o2);
  quadq[rS + j] = f32_to_bf16_bits(o1); quadq[rS + j + 64] = f32_to_bf16_bits(o2);
  APPLY_ROPE(qkw, qkb, o1, o2);
  quadk[rS + j] = f32_to_bf16_bits(o1); quadk[rS + j + 64] = f32_to_bf16_bits(o2);
  APPLY_ROPE(lqw, lqb, o1, o2);
  linq[rS + j] = f32_to_bf16_bits(o1); linq[rS + j + 64] = f32_to_bf16_bits(o2);
  APPLY_ROPE(lkw, lkb, o1, o2);
  size_t tbase = ((size_t)(bb * NCH + g)) * SDIM * CHUNKW;
  linkT[tbase + (size_t)j * CHUNKW + nn] = f32_to_bf16_bits(o1);
  linkT[tbase + (size_t)(j + 64) * CHUNKW + nn] = f32_to_bf16_bits(o2);
#undef APPLY_ROPE
}

// ---------------- per-chunk lin_kv = lin_k^T @ V / 64 -> transposed [B][nc][EH][S] bf16 ----------------
__global__ void linkv_kernel(const unsigned short* __restrict__ linkT, // [B][nc][S][64]
                             const unsigned short* __restrict__ vT,    // [B][EH][LP]
                             unsigned short* __restrict__ kvT) {       // [B][nc][EH][S]
  __shared__ __align__(16) unsigned short As[SDIM * CHUNKW]; // 16KB lin_k^T tile
  int bg = blockIdx.x, bb = bg / NCH, g = bg % NCH;
  int lane = threadIdx.x & 31, wave = threadIdx.x >> 5;
  if (wave == 0) {
    tdm_copy_1d_to_lds(linkT + (size_t)bg * SDIM * CHUNKW, As, SDIM * CHUNKW * 2);
    __builtin_amdgcn_s_wait_tensorcnt(0);
  }
  __syncthreads();
  for (int nt = wave; nt < EHDIM / 16; nt += 8) {
    int ncol0 = nt * 16;
    int n = ncol0 + (lane & 15);
    const unsigned short* vp = vT + ((size_t)bb * EHDIM + n) * LP + g * CHUNKW;
    int kb = (lane >> 4) << 4;
    FragAB fb0, fb1;
    fb0.q[0] = *(const u32x4*)(vp + kb);      fb0.q[1] = *(const u32x4*)(vp + kb + 8);
    fb1.q[0] = *(const u32x4*)(vp + 32 + kb); fb1.q[1] = *(const u32x4*)(vp + 32 + kb + 8);
#pragma unroll
    for (int mt = 0; mt < 8; ++mt) {
      f32x8 acc = {};
      acc = wmma_bf16(load_A_frag(As, CHUNKW, mt * 16, 0, lane), fb0.v, acc);
      acc = wmma_bf16(load_A_frag(As, CHUNKW, mt * 16, 32, lane), fb1.v, acc);
      unsigned short* dp = kvT + ((size_t)bg * EHDIM + n) * SDIM + mt * 16 + ((lane >> 4) << 3);
#pragma unroll
      for (int r = 0; r < 8; ++r) dp[r] = f32_to_bf16_bits(acc[r] * (1.0f / 64.0f));
    }
  }
}

// ---------------- exclusive cumsum over chunks: cumT[b][g][e][s] = sum_{h<g} kvT[b][h][e][s] ----------------
__global__ void cumsum_kernel(const unsigned short* __restrict__ kvT, unsigned short* __restrict__ cumT) {
  size_t i = (size_t)blockIdx.x * blockDim.x + threadIdx.x;
  const size_t per = (size_t)EHDIM * SDIM;
  if (i >= (size_t)BATCH * per) return;
  int bb = (int)(i / per);
  size_t off = i % per;
  float acc = 0.f;
  for (int g = 0; g < NCH; ++g) {
    size_t idx = ((size_t)(bb * NCH + g)) * per + off;
    cumT[idx] = f32_to_bf16_bits(acc);
    acc += bf16_bits_to_f32(kvT[idx]);
  }
}

// ---------------- fused per-chunk: relu^2(QK^T/64+bias) -> LDS; (kernel@V + lin_q@cum) * u -> xh bf16 ----------------
__global__ void chunk_attn_kernel(const unsigned short* __restrict__ quadq,
                                  const unsigned short* __restrict__ quadk,
                                  const unsigned short* __restrict__ linq,
                                  const unsigned short* __restrict__ vT,
                                  const unsigned short* __restrict__ cumT,
                                  const unsigned short* __restrict__ uT,
                                  const float* __restrict__ wrel,
                                  unsigned short* __restrict__ xh) {
  __shared__ __align__(16) unsigned short ker[64 * 64];        // 8KB kernel matrix
  __shared__ __align__(16) unsigned short qs[CHUNKW * SDIM];   // 16KB quad_q tile
  __shared__ __align__(16) unsigned short ks[CHUNKW * SDIM];   // 16KB quad_k tile
  __shared__ __align__(16) unsigned short lqs[CHUNKW * SDIM];  // 16KB lin_q tile
  int bg = blockIdx.x, bb = bg / NCH, g = bg % NCH;
  int lane = threadIdx.x & 31, wave = threadIdx.x >> 5;
  size_t tileoff = ((size_t)(bb * LP + g * CHUNKW)) * SDIM;
  // stage Q/K/lin_q tiles (contiguous 16KB each) into LDS via the Tensor Data Mover
  if (wave == 0) {
    tdm_copy_1d_to_lds(quadq + tileoff, qs, CHUNKW * SDIM * 2);
    tdm_copy_1d_to_lds(quadk + tileoff, ks, CHUNKW * SDIM * 2);
    tdm_copy_1d_to_lds(linq + tileoff, lqs, CHUNKW * SDIM * 2);
    __builtin_amdgcn_s_wait_tensorcnt(0);
  }
  __syncthreads();
  // phase 1: QK^T 64x64, 16 tiles -> 2 per wave
#pragma unroll
  for (int i = 0; i < 2; ++i) {
    int t = wave * 2 + i, mi = t >> 2, ni = t & 3;
    f32x8 acc = {};
    for (int k = 0; k < SDIM; k += 32)
      acc = wmma_bf16(load_A_frag(qs, SDIM, mi * 16, k, lane),
                      load_B_frag(ks, SDIM, ni * 16, k, lane), acc);
    int mk = ni * 16 + (lane & 15);
#pragma unroll
    for (int r = 0; r < 8; ++r) {
      int mq = mi * 16 + ((lane >> 4) << 3) + r;
      float val = 0.f;
      if (mk <= mq) {
        float x = acc[r] * (1.0f / 64.0f) + wrel[511 + mk - mq];
        x = fmaxf(x, 0.f);
        val = x * x;
      }
      ker[mq * 64 + mk] = f32_to_bf16_bits(val);
    }
  }
  __syncthreads();
  // phase 2: per 16-wide EH column strip
  for (int nt = wave; nt < EHDIM / 16; nt += 8) {
    int n = nt * 16 + (lane & 15);
    f32x8 acc[4] = {};
    int kb = (lane >> 4) << 4;
    // quadratic: kernel(64x64) @ V(64x16)
    const unsigned short* vp = vT + ((size_t)bb * EHDIM + n) * LP + g * CHUNKW;
    for (int kk = 0; kk < 64; kk += 32) {
      FragAB fb;
      fb.q[0] = *(const u32x4*)(vp + kk + kb);
      fb.q[1] = *(const u32x4*)(vp + kk + kb + 8);
#pragma unroll
      for (int mt = 0; mt < 4; ++mt)
        acc[mt] = wmma_bf16(load_A_frag(ker, 64, mt * 16, kk, lane), fb.v, acc[mt]);
    }
    // linear: lin_q(64x128) @ cum(128x16)
    const unsigned short* cp = cumT + ((size_t)bg * EHDIM + n) * SDIM;
    for (int kk = 0; kk < SDIM; kk += 32) {
      FragAB fb;
      fb.q[0] = *(const u32x4*)(cp + kk + kb);
      fb.q[1] = *(const u32x4*)(cp + kk + kb + 8);
#pragma unroll
      for (int mt = 0; mt < 4; ++mt)
        acc[mt] = wmma_bf16(load_A_frag(lqs, SDIM, mt * 16, kk, lane), fb.v, acc[mt]);
    }
    // gate by u, store xh row-major bf16
    const unsigned short* up = uT + ((size_t)bb * EHDIM + n) * LP + g * CHUNKW;
#pragma unroll
    for (int mt = 0; mt < 4; ++mt) {
      int mbase = mt * 16 + ((lane >> 4) << 3);
#pragma unroll
      for (int r = 0; r < 8; ++r) {
        int m = mbase + r;
        float y = bf16_bits_to_f32(up[m]) * acc[mt][r];
        xh[((size_t)(bb * LP + g * CHUNKW + m)) * EHDIM + n] = f32_to_bf16_bits(y);
      }
    }
  }
}

// ---------------- final: out[l,b,e] = xh @ Wo^T + bo + query ----------------
__global__ void gemm_out_kernel(const unsigned short* __restrict__ xh,  // [ROWS][EH]
                                const unsigned short* __restrict__ woT, // [E][EH]
                                const float* __restrict__ bo, const float* __restrict__ query,
                                float* __restrict__ out) {
  int lane = threadIdx.x & 31, wave = threadIdx.x >> 5;
  int m0 = blockIdx.x * 64;
  int n0 = blockIdx.y * 128 + wave * 16;
  f32x8 acc[4] = {};
  for (int k = 0; k < EHDIM; k += 32) {
    bf16x16 bfr = load_B_frag(woT, EHDIM, n0, k, lane);
    __builtin_prefetch(woT + (size_t)(n0 + (lane & 15)) * EHDIM + k + 128, 0, 0);
#pragma unroll
    for (int mt = 0; mt < 4; ++mt)
      acc[mt] = wmma_bf16(load_A_frag(xh, EHDIM, m0 + mt * 16, k, lane), bfr, acc[mt]);
  }
  int n = n0 + (lane & 15);
  float bs = bo[n];
#pragma unroll
  for (int mt = 0; mt < 4; ++mt) {
#pragma unroll
    for (int r = 0; r < 8; ++r) {
      int m = m0 + mt * 16 + ((lane >> 4) << 3) + r;
      int bb = m / LP, l = m % LP;
      size_t oi = ((size_t)l * BATCH + bb) * EDIM + n;
      out[oi] = acc[mt][r] + bs + query[oi];
    }
  }
}

extern "C" void kernel_launch(void* const* d_in, const int* in_sizes, int n_in,
                              void* d_out, int out_size, void* d_ws, size_t ws_size,
                              hipStream_t stream) {
  (void)in_sizes; (void)n_in; (void)out_size; (void)ws_size;
  const float* query = (const float*)d_in[0];
  const float* ln_g  = (const float*)d_in[3];
  const float* ln_b  = (const float*)d_in[4];
  const float* Wu    = (const float*)d_in[5];
  const float* bu    = (const float*)d_in[6];
  const float* Wv    = (const float*)d_in[7];
  const float* bv    = (const float*)d_in[8];
  const float* Wb    = (const float*)d_in[9];
  const float* bbv   = (const float*)d_in[10];
  const float* qqw   = (const float*)d_in[11];
  const float* qqb   = (const float*)d_in[12];
  const float* qkw   = (const float*)d_in[13];
  const float* qkb   = (const float*)d_in[14];
  const float* lqw   = (const float*)d_in[15];
  const float* lqb   = (const float*)d_in[16];
  const float* lkw   = (const float*)d_in[17];
  const float* lkb   = (const float*)d_in[18];
  const float* Wo    = (const float*)d_in[19];
  const float* bo    = (const float*)d_in[20];
  const float* wrel  = (const float*)d_in[21];
  float* out = (float*)d_out;

  char* ws = (char*)d_ws;
  size_t off = 0;
  auto alloc = [&](size_t bytes) {
    char* p = ws + off;
    off += (bytes + 255) & ~(size_t)255;
    return p;
  };
  unsigned short* wu_bf  = (unsigned short*)alloc((size_t)EHDIM * EDIM * 2);
  unsigned short* wv_bf  = (unsigned short*)alloc((size_t)EHDIM * EDIM * 2);
  unsigned short* wb_bf  = (unsigned short*)alloc((size_t)SDIM * EDIM * 2);
  unsigned short* wo_bf  = (unsigned short*)alloc((size_t)EDIM * EHDIM * 2);
  unsigned short* xn_bf  = (unsigned short*)alloc((size_t)ROWS * EDIM * 2);
  unsigned short* uT_bf  = (unsigned short*)alloc((size_t)BATCH * EHDIM * LP * 2);
  unsigned short* vT_bf  = (unsigned short*)alloc((size_t)BATCH * EHDIM * LP * 2);
  float*          base_f = (float*)alloc((size_t)ROWS * SDIM * 4);
  unsigned short* qq_bf  = (unsigned short*)alloc((size_t)ROWS * SDIM * 2);
  unsigned short* qk_bf  = (unsigned short*)alloc((size_t)ROWS * SDIM * 2);
  unsigned short* lq_bf  = (unsigned short*)alloc((size_t)ROWS * SDIM * 2);
  unsigned short* lkT_bf = (unsigned short*)alloc((size_t)ROWS * SDIM * 2);
  unsigned short* kvT_bf = (unsigned short*)alloc((size_t)BATCH * NCH * EHDIM * SDIM * 2);
  unsigned short* cum_bf = (unsigned short*)alloc((size_t)BATCH * NCH * EHDIM * SDIM * 2);
  unsigned short* xh_bf  = (unsigned short*)alloc((size_t)ROWS * EHDIM * 2);

  cvt_bf16_kernel<<<(EHDIM * EDIM + 255) / 256, 256, 0, stream>>>(Wu, wu_bf, EHDIM * EDIM);
  cvt_bf16_kernel<<<(EHDIM * EDIM + 255) / 256, 256, 0, stream>>>(Wv, wv_bf, EHDIM * EDIM);
  cvt_bf16_kernel<<<(SDIM * EDIM + 255) / 256, 256, 0, stream>>>(Wb, wb_bf, SDIM * EDIM);
  cvt_bf16_kernel<<<(EDIM * EHDIM + 255) / 256, 256, 0, stream>>>(Wo, wo_bf, EDIM * EHDIM);

  ln_kernel<<<ROWS, 256, 0, stream>>>(query, ln_g, ln_b, xn_bf);

  gemm1_kernel<<<dim3(ROWS / 64, EHDIM / 128), 256, 0, stream>>>(xn_bf, wu_bf, bu, EHDIM, 0, uT_bf, nullptr);
  gemm1_kernel<<<dim3(ROWS / 64, EHDIM / 128), 256, 0, stream>>>(xn_bf, wv_bf, bv, EHDIM, 0, vT_bf, nullptr);
  gemm1_kernel<<<dim3(ROWS / 64, 1), 256, 0, stream>>>(xn_bf, wb_bf, bbv, SDIM, 1, nullptr, base_f);

  rope_kernel<<<(ROWS * 64) / 256, 256, 0, stream>>>(base_f, qqw, qqb, qkw, qkb, lqw, lqb, lkw, lkb,
                                                     qq_bf, qk_bf, lq_bf, lkT_bf);

  linkv_kernel<<<BATCH * NCH, 256, 0, stream>>>(lkT_bf, vT_bf, kvT_bf);
  cumsum_kernel<<<(BATCH * EHDIM * SDIM) / 256, 256, 0, stream>>>(kvT_bf, cum_bf);

  chunk_attn_kernel<<<BATCH * NCH, 256, 0, stream>>>(qq_bf, qk_bf, lq_bf, vT_bf, cum_bf, uT_bf, wrel, xh_bf);

  gemm_out_kernel<<<dim3(ROWS / 64, EDIM / 128), 256, 0, stream>>>(xh_bf, wo_bf, bo, query, out);
}
